// OrdinalContrastiveLoss_mm_36266703848098
// MI455X (gfx1250) — matrix operand order repaired and checked by reading, not compiled, not run on hardware
//
#include <hip/hip_runtime.h>
#include <hip/hip_bf16.h>

// ---------------------------------------------------------------------------
// OrdinalContrastiveLoss for MI455X (gfx1250, wave32, WMMA)
//   N = 512 samples, D = 512 emb, C = 5 classes
// ---------------------------------------------------------------------------

#define N_SAMPLES 512
#define D_EMB     512
#define N_CLASSES 5

typedef __attribute__((ext_vector_type(16))) _Float16 v16h;
typedef __attribute__((ext_vector_type(8)))  _Float16 v8h;
typedef __attribute__((ext_vector_type(8)))  float    v8f;

// ---------------------------------------------------------------------------
// Kernel 1: row L2-normalize + split into f16 hi/lo pair (Dekker-style split
// so that the WMMA gram matrix reproduces fp32-accurate cosine similarity).
// One wave (32 lanes) per row; 8 waves / block; 64 blocks.
// ---------------------------------------------------------------------------
__global__ void k_norm_split(const float* __restrict__ X,
                             _Float16* __restrict__ Xhi,
                             _Float16* __restrict__ Xlo) {
    const int row  = blockIdx.x * 8 + (threadIdx.x >> 5);
    const int lane = threadIdx.x & 31;
    const float* xr = X + row * D_EMB;

    float vals[16];
    float ss = 0.0f;
#pragma unroll
    for (int c = 0; c < 16; ++c) {
        float v = xr[lane + 32 * c];
        vals[c] = v;
        ss += v * v;
    }
    // wave32 butterfly reduction
#pragma unroll
    for (int off = 16; off > 0; off >>= 1)
        ss += __shfl_xor(ss, off, 32);

    const float inv = 1.0f / fmaxf(sqrtf(ss), 1e-8f);

#pragma unroll
    for (int c = 0; c < 16; ++c) {
        float xn = vals[c] * inv;
        _Float16 h = (_Float16)xn;
        float    lo = xn - (float)h;
        Xhi[row * D_EMB + lane + 32 * c] = h;
        Xlo[row * D_EMB + lane + 32 * c] = (_Float16)lo;
    }
}

// ---------------------------------------------------------------------------
// Kernel 2: class positions (softplus cumsum), per-sample class position cp[i]
// and negative counts negcnt[i].  One block of 512 threads.
// ---------------------------------------------------------------------------
__global__ void k_meta(const float* __restrict__ draw,
                       const int* __restrict__ tgt,
                       float* __restrict__ cp,
                       int* __restrict__ negcnt) {
    __shared__ float cls[N_CLASSES];
    __shared__ int   cnt[N_CLASSES];
    const int t = threadIdx.x;
    if (t < N_CLASSES) cnt[t] = 0;
    if (t == 0) {
        float acc = 0.0f;
        cls[0] = 0.0f;
#pragma unroll
        for (int c = 0; c < N_CLASSES - 1; ++c) {
            acc += log1pf(expf(draw[c]));   // softplus, then cumsum
            cls[c + 1] = acc;
        }
    }
    __syncthreads();
    const int lab = tgt[t];
    atomicAdd(&cnt[lab], 1);
    __syncthreads();
    cp[t]     = cls[lab];
    negcnt[t] = N_SAMPLES - cnt[lab];
}

// ---------------------------------------------------------------------------
// Kernel 3: Gram matrix cos = Xn * Xn^T via v_wmma_f32_16x16x32_f16.
// One wave per 16x16 output tile; grid (32, 32).
// Two-term split: cos ~= hi*hi + hi*lo + lo*hi  (f32 accumulate).
//
// A layout (16x32 f16, wave32): lane l, m=l&15, holds K in
//   [8*(l>=16), +8) and [16+8*(l>=16), +8)   -> two contiguous 8-half loads.
// B layout (32x16 f16, column-major == rows of X): lane l, n=l&15, holds
//   the same K as A: [8*(l>=16), +8) and [16+8*(l>=16), +8) -> two 8-half loads.
// C/D: VGPR r -> M = r (+8 for lanes 16-31), N = l&15.
// ---------------------------------------------------------------------------
__global__ void k_gram_wmma(const _Float16* __restrict__ Xhi,
                            const _Float16* __restrict__ Xlo,
                            float* __restrict__ cosM) {
    const int tm   = blockIdx.x * 16;
    const int tn   = blockIdx.y * 16;
    const int l    = threadIdx.x;        // 0..31
    const int hi16 = l >> 4;             // 0 or 1
    const int idx  = l & 15;

    const _Float16* arow_hi = Xhi + (size_t)(tm + idx) * D_EMB;
    const _Float16* arow_lo = Xlo + (size_t)(tm + idx) * D_EMB;
    const _Float16* brow_hi = Xhi + (size_t)(tn + idx) * D_EMB;
    const _Float16* brow_lo = Xlo + (size_t)(tn + idx) * D_EMB;

    const int aoff = hi16 * 8;           // start of first 8-half run
    const int boff = hi16 * 8;           // start of first 8-half run

    v8f acc = {};

    for (int k0 = 0; k0 < D_EMB; k0 += 32) {
        union { v16h v; v8h h[2]; } ah, al;
        ah.h[0] = *(const v8h*)(arow_hi + k0 + aoff);
        ah.h[1] = *(const v8h*)(arow_hi + k0 + aoff + 16);
        al.h[0] = *(const v8h*)(arow_lo + k0 + aoff);
        al.h[1] = *(const v8h*)(arow_lo + k0 + aoff + 16);

        v16h bh = __builtin_shufflevector(*(const v8h*)(brow_hi + k0 + boff), *(const v8h*)(brow_hi + k0 + boff + 16), 0,1,2,3,4,5,6,7,8,9,10,11,12,13,14,15);
        v16h bl = __builtin_shufflevector(*(const v8h*)(brow_lo + k0 + boff), *(const v8h*)(brow_lo + k0 + boff + 16), 0,1,2,3,4,5,6,7,8,9,10,11,12,13,14,15);

        // cos += hi*hi + hi*lo + lo*hi   (drop lo*lo: ~2^-22 relative)
        acc = __builtin_amdgcn_wmma_f32_16x16x32_f16(false, ah.v, false, bh,
                                                     (short)0, acc, false, false);
        acc = __builtin_amdgcn_wmma_f32_16x16x32_f16(false, ah.v, false, bl,
                                                     (short)0, acc, false, false);
        acc = __builtin_amdgcn_wmma_f32_16x16x32_f16(false, al.v, false, bh,
                                                     (short)0, acc, false, false);
    }

#pragma unroll
    for (int r = 0; r < 8; ++r) {
        const int row = tm + r + hi16 * 8;
        const int col = tn + idx;
        cosM[(size_t)row * N_SAMPLES + col] = acc[r];
    }
}

// ---------------------------------------------------------------------------
// Kernel 4: per-anchor hinge accumulation.  One block (256 threads) per
// anchor i.  bL[j] holds cos[i,j]+margin for negatives, -1e30 sentinel for
// non-negatives so the inner loop is a branch-free relu-sum.
// ---------------------------------------------------------------------------
__global__ void k_pairloss(const float* __restrict__ cosM,
                           const float* __restrict__ cp,
                           const int* __restrict__ tgt,
                           const int* __restrict__ negcnt,
                           float* __restrict__ rowSum,
                           int* __restrict__ rowCnt) {
    __shared__ float crow[N_SAMPLES];
    __shared__ float scp [N_SAMPLES];
    __shared__ int   stgt[N_SAMPLES];
    __shared__ float bL  [N_SAMPLES];
    __shared__ float redS[256];
    __shared__ int   redC[256];

    const int i = blockIdx.x;
    const int t = threadIdx.x;

    for (int j = t; j < N_SAMPLES; j += 256) {
        crow[j] = cosM[(size_t)i * N_SAMPLES + j];
        scp[j]  = cp[j];
        stgt[j] = tgt[j];
    }
    __syncthreads();

    const int   ti  = stgt[i];
    const float cpi = scp[i];

    for (int j = t; j < N_SAMPLES; j += 256) {
        const bool neg = (stgt[j] != ti);
        bL[j] = neg ? (crow[j] + fabsf(cpi - scp[j])) : -1e30f;
    }
    __syncthreads();

    const int   nc   = negcnt[i];
    const float invn = 1.0f / (float)(nc > 1 ? nc : 1);

    float mySum = 0.0f;
    int   myCnt = 0;
    for (int k = t; k < N_SAMPLES; k += 256) {
        if (stgt[k] == ti && k != i) {          // positive pair (i, k)
            const float th = crow[k];
            float s = 0.0f;
#pragma unroll 8
            for (int j = 0; j < N_SAMPLES; ++j)
                s += fmaxf(bL[j] - th, 0.0f);   // sentinel -> exact 0
            if (s > 0.0f) { mySum += s * invn; myCnt += 1; }
        }
    }

    redS[t] = mySum;
    redC[t] = myCnt;
    __syncthreads();
#pragma unroll
    for (int off = 128; off > 0; off >>= 1) {
        if (t < off) { redS[t] += redS[t + off]; redC[t] += redC[t + off]; }
        __syncthreads();
    }
    if (t == 0) { rowSum[i] = redS[0]; rowCnt[i] = redC[0]; }
}

// ---------------------------------------------------------------------------
// Kernel 5: final deterministic reduction to the scalar loss.
// ---------------------------------------------------------------------------
__global__ void k_finalize(const float* __restrict__ rowSum,
                           const int* __restrict__ rowCnt,
                           float* __restrict__ out) {
    __shared__ float s[N_SAMPLES];
    __shared__ int   c[N_SAMPLES];
    const int t = threadIdx.x;
    s[t] = rowSum[t];
    c[t] = rowCnt[t];
    __syncthreads();
#pragma unroll
    for (int off = 256; off > 0; off >>= 1) {
        if (t < off) { s[t] += s[t + off]; c[t] += c[t + off]; }
        __syncthreads();
    }
    if (t == 0) out[0] = s[0] / (float)(c[0] > 1 ? c[0] : 1);
}

// ---------------------------------------------------------------------------
// Launcher
// ---------------------------------------------------------------------------
extern "C" void kernel_launch(void* const* d_in, const int* in_sizes, int n_in,
                              void* d_out, int out_size, void* d_ws, size_t ws_size,
                              hipStream_t stream) {
    const float* prediction    = (const float*)d_in[0];   // [512, 512] f32
    const float* distances_raw = (const float*)d_in[1];   // [4] f32
    const int*   target        = (const int*)d_in[2];     // [512] i32
    float*       out           = (float*)d_out;           // scalar f32

    // workspace layout (bytes)
    char* ws = (char*)d_ws;
    float*     cosM   = (float*)    (ws);                                  // 1 MB
    _Float16*  Xhi    = (_Float16*) (ws + 1048576);                        // 512 KB
    _Float16*  Xlo    = (_Float16*) (ws + 1048576 + 524288);               // 512 KB
    float*     cp     = (float*)    (ws + 2097152);                        // 2 KB
    int*       negcnt = (int*)      (ws + 2097152 + 2048);                 // 2 KB
    float*     rowSum = (float*)    (ws + 2097152 + 4096);                 // 2 KB
    int*       rowCnt = (int*)      (ws + 2097152 + 6144);                 // 2 KB

    // 1) normalize rows + f16 hi/lo split
    k_norm_split<<<dim3(N_SAMPLES / 8), dim3(256), 0, stream>>>(prediction, Xhi, Xlo);

    // 2) class positions + negative counts
    k_meta<<<dim3(1), dim3(N_SAMPLES), 0, stream>>>(distances_raw, target, cp, negcnt);

    // 3) cosine Gram matrix via WMMA (one wave per 16x16 tile)
    k_gram_wmma<<<dim3(N_SAMPLES / 16, N_SAMPLES / 16), dim3(32), 0, stream>>>(Xhi, Xlo, cosM);

    // 4) per-anchor hinge sums
    k_pairloss<<<dim3(N_SAMPLES), dim3(256), 0, stream>>>(cosM, cp, target, negcnt,
                                                          rowSum, rowCnt);

    // 5) scalar reduction
    k_finalize<<<dim3(1), dim3(N_SAMPLES), 0, stream>>>(rowSum, rowCnt, out);
}
